// G_VQVAE_28767690948720
// MI455X (gfx1250) — compile-verified, matrix operand order checked by reading, not run on hardware
//
#include <hip/hip_runtime.h>

#define NN   50000
#define NE   800000
#define CIN  128
#define HID  256
#define KCB  512
#define DD   256
#define BETA 0.25f

typedef __attribute__((ext_vector_type(16))) __bf16 v16bf;
typedef __attribute__((ext_vector_type(8)))  float  v8f;

static __device__ __forceinline__ unsigned short f2bf(float f) {
  unsigned u = __float_as_uint(f);
  unsigned r = u + 0x7FFFu + ((u >> 16) & 1u);   // round-to-nearest-even
  return (unsigned short)(r >> 16);
}

// Async copy of one 16-byte chunk: global -> LDS (gfx1250 ASYNCcnt path).
static __device__ __forceinline__ void async_g2l_b128(unsigned lds_off,
                                                      unsigned long long gaddr) {
  asm volatile("global_load_async_to_lds_b128 %0, %1, off"
               :: "v"(lds_off), "v"(gaddr) : "memory");
}
static __device__ __forceinline__ void wait_async0() {
  asm volatile("s_wait_asynccnt 0" ::: "memory");
}

// ---------------- basic utility kernels ----------------

__global__ void k_fill(float* p, float v, int n) {
  int i = blockIdx.x * blockDim.x + threadIdx.x;
  if (i < n) p[i] = v;
}

__global__ void k_deg(const int* dst, float* deg, int e) {
  int i = blockIdx.x * blockDim.x + threadIdx.x;
  if (i < e) atomicAdd(&deg[dst[i]], 1.0f);
}

__global__ void k_dinv(const float* deg, float* dinv, float* deginv, int n) {
  int i = blockIdx.x * blockDim.x + threadIdx.x;
  if (i < n) { float d = deg[i]; dinv[i] = rsqrtf(d); deginv[i] = 1.0f / d; }
}

// ---------------- WMMA operand packing ----------------
// Per (strip, ktile): a 32-lane x 16-bf16 block laid out exactly as the v16bf
// register image (contiguous 32B per lane).  Element e of lane L maps to
// K = ktile*32 + (e>>3)*16 + (L>>4)*8 + ((e>>1)&3)*2 + (e&1); row/col = (L&15).

__global__ void k_pack_a(const float* __restrict__ in, unsigned short* __restrict__ out,
                         int K, int strips) {
  long t = blockIdx.x * (long)blockDim.x + threadIdx.x;
  int KT = K >> 5;
  long total = (long)strips * KT * 512;
  if (t >= total) return;
  int e = (int)(t & 15);
  int lane = (int)((t >> 4) & 31);
  int kt = (int)((t >> 9) % KT);
  int strip = (int)(t / (512L * KT));
  int kidx = kt * 32 + (e >> 3) * 16 + (lane >> 4) * 8 + ((e >> 1) & 3) * 2 + (e & 1);
  int row = strip * 16 + (lane & 15);
  out[t] = f2bf(in[(long)row * K + kidx]);
}

__global__ void k_pack_b(const float* __restrict__ W, unsigned short* __restrict__ out,
                         int K, int Nout, int transposed) {
  long t = blockIdx.x * (long)blockDim.x + threadIdx.x;
  int KT = K >> 5;
  int ntiles = Nout >> 4;
  long total = (long)ntiles * KT * 512;
  if (t >= total) return;
  int e = (int)(t & 15);
  int lane = (int)((t >> 4) & 31);
  int kt = (int)((t >> 9) % KT);
  int nt = (int)(t / (512L * KT));
  int kidx = kt * 32 + (e >> 3) * 16 + (lane >> 4) * 8 + ((e >> 1) & 3) * 2 + (e & 1);
  int col = nt * 16 + (lane & 15);
  float v = transposed ? W[(long)col * K + kidx]       // B = W^T (codebook)
                       : W[(long)kidx * Nout + col];   // B = W (row-major [K,Nout])
  out[t] = f2bf(v);
}

// ---------------- bf16 WMMA GEMM with async LDS staging ----------------
// Block = 4 waves; A strip (KT KB) + 4 B tiles (4*KT KB) async-copied to LDS,
// then each wave runs a 16x16x32 WMMA K-loop out of ds_load_b128.

__global__ void __launch_bounds__(128)
k_gemm(const unsigned short* __restrict__ Ap_, const unsigned short* __restrict__ Bp_,
       float* __restrict__ out, const float* __restrict__ bias,
       int KT, int Nout, int relu) {
  __shared__ unsigned char sAB[40 * 1024];   // [0,8K) = A strip, [8K,40K) = B tiles
  int tid = threadIdx.x;
  int lane = tid & 31;
  int wave = tid >> 5;
  int strip = blockIdx.x;

  // async stage A strip: KT*1024 bytes contiguous
  {
    const unsigned char* gA = (const unsigned char*)(Ap_ + (size_t)strip * KT * 512);
    unsigned baseA = (unsigned)(size_t)&sAB[0];
    int nchunks = KT * 64;                       // 16B chunks
    for (int c = tid; c < nchunks; c += 128)
      async_g2l_b128(baseA + c * 16, (unsigned long long)(gA + c * 16));
  }
  // async stage B tiles for this block's 4 ntiles: 4*KT*1024 bytes contiguous
  {
    const unsigned char* gB = (const unsigned char*)(Bp_ + (size_t)blockIdx.y * 4 * KT * 512);
    unsigned baseB = (unsigned)(size_t)&sAB[8192];
    int nchunks = KT * 256;
    for (int c = tid; c < nchunks; c += 128)
      async_g2l_b128(baseB + c * 16, (unsigned long long)(gB + c * 16));
  }
  wait_async0();
  __syncthreads();

  const v16bf* sA = (const v16bf*)&sAB[0];
  const v16bf* sB = (const v16bf*)&sAB[8192 + wave * KT * 1024];
  v8f c = {};
  for (int kt = 0; kt < KT; ++kt) {
    v16bf a = sA[kt * 32 + lane];
    v16bf b = sB[kt * 32 + lane];
    c = __builtin_amdgcn_wmma_f32_16x16x32_bf16(false, a, false, b, (short)0, c, false, false);
  }
  int nt = blockIdx.y * 4 + wave;
  int col = nt * 16 + (lane & 15);
  float bv = bias ? bias[col] : 0.0f;
  int mb = (lane >> 4) * 8;
  for (int r = 0; r < 8; ++r) {
    int row = strip * 16 + mb + r;
    float v = c[r] + bv;
    if (relu) v = fmaxf(v, 0.0f);
    out[(long)row * Nout + col] = v;
  }
}

// ---------------- GCN aggregation (Cout = 256 fixed) ----------------

__global__ void k_gcn_init(const float* __restrict__ hW, const float* __restrict__ deginv,
                           float* __restrict__ agg, int total) {
  int i = blockIdx.x * blockDim.x + threadIdx.x;
  if (i < total) agg[i] = hW[i] * deginv[i >> 8];   // self-loop term h/deg
}

__global__ void k_scatter(const int* __restrict__ src, const int* __restrict__ dst,
                          const float* __restrict__ dinv, const float* __restrict__ hW,
                          float* __restrict__ agg) {
  int t = blockIdx.x * blockDim.x + threadIdx.x;   // NE*64 threads, 4 channels each
  int e = t >> 6, cg = t & 63;
  if (e >= NE) return;
  int s = src[e], d = dst[e];
  float w = dinv[s] * dinv[d];
  float4 v = ((const float4*)(hW + (long)s * 256))[cg];
  float* ag = agg + (long)d * 256 + cg * 4;
  atomicAdd(ag + 0, v.x * w);
  atomicAdd(ag + 1, v.y * w);
  atomicAdd(ag + 2, v.z * w);
  atomicAdd(ag + 3, v.w * w);
}

__global__ void k_gcn_final(const float* __restrict__ agg, const float* __restrict__ bias,
                            float* __restrict__ out, int total, int relu) {
  int i = blockIdx.x * blockDim.x + threadIdx.x;
  if (i >= total) return;
  float v = agg[i] + bias[i & 255];
  if (relu) v = fmaxf(v, 0.0f);
  out[i] = v;
}

// ---------------- VQ ----------------

__global__ void k_cnorm(const float* __restrict__ cb, float* __restrict__ cn) {
  int k = blockIdx.x * blockDim.x + threadIdx.x;
  if (k >= KCB) return;
  float s = 0.0f;
  const float* r = cb + (long)k * DD;
  for (int d = 0; d < DD; ++d) s += r[d] * r[d];
  cn[k] = s;
}

// One 16-node strip per block: async-stage the z strip to LDS, WMMA builds the
// 16x512 score tile in LDS, then 16 lanes argmin over ||c||^2 - 2*score.
__global__ void __launch_bounds__(128)
k_vq(const unsigned short* __restrict__ Zp_, const unsigned short* __restrict__ Bp_,
     const float* __restrict__ cnorm, int* __restrict__ idxb, float* __restrict__ idxf) {
  __shared__ float sc[16 * 512];               // 32KB score tile
  __shared__ unsigned char sZ[8192];           // 8KB z strip (KT=8)
  const v16bf* Bp = (const v16bf*)Bp_;
  int tid = threadIdx.x;
  int lane = tid & 31;
  int wave = tid >> 5;
  int strip = blockIdx.x;

  {
    const unsigned char* gZ = (const unsigned char*)(Zp_ + (size_t)strip * 8 * 512);
    unsigned baseZ = (unsigned)(size_t)&sZ[0];
    for (int c = tid; c < 512; c += 128)       // 8KB in 16B chunks
      async_g2l_b128(baseZ + c * 16, (unsigned long long)(gZ + c * 16));
  }
  wait_async0();
  __syncthreads();

  const v16bf* sZv = (const v16bf*)&sZ[0];
  for (int i = 0; i < 8; ++i) {                // 4 waves x 8 = 32 ntiles = 512 codes
    int nt = wave * 8 + i;
    v8f c = {};
    for (int kt = 0; kt < 8; ++kt) {           // K = 256 contraction
      v16bf a = sZv[kt * 32 + lane];
      v16bf b = Bp[((long)nt * 8 + kt) * 32 + lane];
      c = __builtin_amdgcn_wmma_f32_16x16x32_bf16(false, a, false, b, (short)0, c, false, false);
    }
    int col = nt * 16 + (lane & 15);
    int mb = (lane >> 4) * 8;
    for (int r = 0; r < 8; ++r) sc[(mb + r) * 512 + col] = c[r];
  }
  __syncthreads();
  if (tid < 16) {
    int row = tid;
    float best = 3.4e38f;
    int bi = 0;
    for (int k = 0; k < KCB; ++k) {
      float d = cnorm[k] - 2.0f * sc[row * 512 + k];
      if (d < best) { best = d; bi = k; }
    }
    int node = strip * 16 + row;
    idxb[node] = bi;
    idxf[node] = (float)bi;
  }
}

// q = codebook[idx]; accumulate sum((q - z)^2)
__global__ void k_qloss(const float* __restrict__ z, const float* __restrict__ cb,
                        const int* __restrict__ idxb, float* __restrict__ q,
                        float* __restrict__ acc) {
  __shared__ float red[256];
  int n = blockIdx.x, t = threadIdx.x;
  int idx = idxb[n];
  float qv = cb[(long)idx * DD + t];
  float zv = z[(long)n * DD + t];
  q[(long)n * DD + t] = qv;
  float d = qv - zv;
  red[t] = d * d;
  __syncthreads();
  for (int s = 128; s > 0; s >>= 1) {
    if (t < s) red[t] += red[t + s];
    __syncthreads();
  }
  if (t == 0) atomicAdd(acc, red[0]);
}

__global__ void k_lossfin(const float* __restrict__ acc, float* __restrict__ out) {
  if (blockIdx.x == 0 && threadIdx.x == 0)
    out[0] = (1.0f + BETA) * acc[0] / ((float)NN * (float)DD);
}

// ---------------- launcher ----------------

extern "C" void kernel_launch(void* const* d_in, const int* in_sizes, int n_in,
                              void* d_out, int out_size, void* d_ws, size_t ws_size,
                              hipStream_t stream) {
  const float* x    = (const float*)d_in[0];
  const int*   ei   = (const int*)d_in[1];
  const float* W1e  = (const float*)d_in[2];
  const float* b1e  = (const float*)d_in[3];
  const float* W2e  = (const float*)d_in[4];
  const float* b2e  = (const float*)d_in[5];
  const float* Wc   = (const float*)d_in[6];
  const float* bc   = (const float*)d_in[7];
  const float* cb   = (const float*)d_in[8];
  const float* W1d  = (const float*)d_in[9];
  const float* b1d  = (const float*)d_in[10];
  const float* W2d  = (const float*)d_in[11];
  const float* b2d  = (const float*)d_in[12];
  const float* Wf   = (const float*)d_in[13];
  const float* bf   = (const float*)d_in[14];

  const int* src = ei;
  const int* dst = ei + NE;

  float* xrec    = (float*)d_out;            // [NN, CIN]
  float* lossout = xrec + (long)NN * CIN;    // [1]
  float* idxout  = lossout + 1;              // [NN]

  // workspace carve-up (256B aligned)
  char* cur = (char*)d_ws;
  auto alloc = [&](size_t bytes) -> void* {
    void* p = (void*)cur;
    cur += (bytes + 255) & ~((size_t)255);
    return p;
  };
  float* deg    = (float*)alloc((size_t)NN * 4);
  float* dinv   = (float*)alloc((size_t)NN * 4);
  float* deginv = (float*)alloc((size_t)NN * 4);
  int*   idxb   = (int*)alloc((size_t)NN * 4);
  float* cnorm  = (float*)alloc((size_t)KCB * 4);
  float* lacc   = (float*)alloc(256);
  unsigned short* Ap = (unsigned short*)alloc((size_t)3125 * 8 * 1024);  // 25.6 MB (K<=256)
  unsigned short* Wp = (unsigned short*)alloc((size_t)8 * 32 * 1024);    // 256 KB  (Nout<=512)
  float* hW   = (float*)alloc((size_t)NN * HID * 4);
  float* agg  = (float*)alloc((size_t)NN * HID * 4);
  float* bufH = (float*)alloc((size_t)NN * HID * 4);
  float* zbuf = (float*)alloc((size_t)NN * HID * 4);
  float* qbuf = (float*)alloc((size_t)NN * HID * 4);

  const int STRIPS = NN / 16;   // 3125

  // degrees + symmetric norm
  k_fill<<<(NN + 255) / 256, 256, 0, stream>>>(deg, 1.0f, NN);
  k_deg<<<(NE + 255) / 256, 256, 0, stream>>>(dst, deg, NE);
  k_dinv<<<(NN + 255) / 256, 256, 0, stream>>>(deg, dinv, deginv, NN);

  auto dense = [&](const float* in, int K, const float* W, const float* b,
                   int Nout, int transposed, int relu, float* outp) {
    int KT = K >> 5, ntiles = Nout >> 4;
    long tb = (long)ntiles * KT * 512;
    k_pack_b<<<(int)((tb + 255) / 256), 256, 0, stream>>>(W, Wp, K, Nout, transposed);
    long ta = (long)STRIPS * KT * 512;
    k_pack_a<<<(int)((ta + 255) / 256), 256, 0, stream>>>(in, Ap, K, STRIPS);
    k_gemm<<<dim3(STRIPS, ntiles / 4), 128, 0, stream>>>(Ap, Wp, outp, b, KT, Nout, relu);
  };

  auto gcn = [&](const float* in, int K, const float* W, const float* b, float* outp) {
    dense(in, K, W, nullptr, HID, 0, 0, hW);           // hW = X W
    int tot = NN * HID;
    k_gcn_init<<<(tot + 255) / 256, 256, 0, stream>>>(hW, deginv, agg, tot);
    k_scatter<<<(NE * 64) / 256, 256, 0, stream>>>(src, dst, dinv, hW, agg);
    k_gcn_final<<<(tot + 255) / 256, 256, 0, stream>>>(agg, b, outp, tot, 1);
  };

  // encoder
  gcn(x, CIN, W1e, b1e, bufH);
  gcn(bufH, HID, W2e, b2e, bufH);
  dense(bufH, HID, Wc, bc, DD, 0, 0, zbuf);            // z = h Wc + bc

  // VQ
  k_cnorm<<<(KCB + 255) / 256, 256, 0, stream>>>(cb, cnorm);
  {
    long ta = (long)STRIPS * 8 * 512;
    k_pack_a<<<(int)((ta + 255) / 256), 256, 0, stream>>>(zbuf, Ap, DD, STRIPS);
    long tb = (long)(KCB >> 4) * 8 * 512;
    k_pack_b<<<(int)((tb + 255) / 256), 256, 0, stream>>>(cb, Wp, DD, KCB, 1);
    k_vq<<<STRIPS, 128, 0, stream>>>(Ap, Wp, cnorm, idxb, idxout);
  }
  k_fill<<<1, 256, 0, stream>>>(lacc, 0.0f, 1);
  k_qloss<<<NN, 256, 0, stream>>>(zbuf, cb, idxb, qbuf, lacc);
  k_lossfin<<<1, 1, 0, stream>>>(lacc, lossout);

  // decoder
  gcn(qbuf, DD, W1d, b1d, bufH);
  gcn(bufH, HID, W2d, b2d, bufH);
  dense(bufH, HID, Wf, bf, CIN, 0, 0, xrec);           // x_recon
}
